// DecisionTransformer_82738249990831
// MI455X (gfx1250) — compile-verified, matrix operand order checked by reading, not run on hardware
//
#include <hip/hip_runtime.h>
#include <hip/hip_bf16.h>

typedef _Float16 half_t;
typedef __attribute__((ext_vector_type(16))) _Float16 v16h;
typedef __attribute__((ext_vector_type(8)))  float    v8f;

constexpr int NB   = 4;     // batches
constexpr int LT   = 512;   // timesteps
constexpr int DM   = 768;   // model dim
constexpr int SQ   = 3 * LT;     // 1536 tokens
constexpr int FFD  = 2048;
constexpr int NLAY = 4;
constexpr int NH   = 12;
constexpr int HD   = 64;    // head dim (KD == VD)
constexpr int ACT  = 6;
constexpr int STATE = 17;

// ---- WMMA f16 fragment helpers (16x16x32, wave32) -------------------------
// A fragment (16xK rows, also used for B as N x K rows, C = A * B^T):
// lane m = lane%16 is the row, lane group g = lane/16 picks the K interleave.
// VGPR j holds half-pair starting at kmap(j,g) (ISA 7.12.2 16-bit A layout).
__device__ __forceinline__ int kmap(int j, int g) {
  return (j < 4 ? 2 * j : 8 + 2 * j) + (g ? 8 : 0);
}

__device__ __forceinline__ v16h load_frag(const half_t* base, int ld) {
  union { v16h v; unsigned u[8]; } f;
  int lane = threadIdx.x & 31;
  int m = lane & 15, g = lane >> 4;
  const char* row = (const char*)(base + (size_t)m * ld);
#pragma unroll
  for (int j = 0; j < 8; ++j)
    f.u[j] = *(const unsigned*)(row + 2 * kmap(j, g));
  return f.v;
}

__device__ __forceinline__ v8f wmma32(v16h a, v16h b, v8f c) {
  return __builtin_amdgcn_wmma_f32_16x16x32_f16(false, a, false, b, (short)0, c,
                                                false, false);
}

// ---- block reduction (256 threads = 8 waves) ------------------------------
__device__ __forceinline__ float block_sum(float v, float* sbuf) {
  int lane = threadIdx.x & 31, w = threadIdx.x >> 5;
#pragma unroll
  for (int m = 16; m >= 1; m >>= 1) v += __shfl_xor(v, m, 32);
  if (lane == 0) sbuf[w] = v;
  __syncthreads();
  float r = (lane < 8) ? sbuf[lane] : 0.0f;
#pragma unroll
  for (int m = 4; m >= 1; m >>= 1) r += __shfl_xor(r, m, 32);
  if (threadIdx.x == 0) sbuf[0] = r;
  __syncthreads();
  float out = sbuf[0];
  __syncthreads();
  return out;
}

// ---- weight convert + transpose: src f32 [K, Nout] -> dst f16 [Nout, K] ---
__global__ void conv_transpose_f16(const float* __restrict__ src,
                                   half_t* __restrict__ dst, int K, int Nout) {
  int idx = blockIdx.x * blockDim.x + threadIdx.x;
  if (idx >= K * Nout) return;
  int k = idx / Nout, n = idx - k * Nout;
  dst[(size_t)n * K + k] = (half_t)src[idx];
}

// ---- embedding + interleave + LN0 -----------------------------------------
__global__ void embed_ln0(const float* __restrict__ R, const float* __restrict__ sv,
                          const float* __restrict__ av, const int* __restrict__ tt,
                          const float* __restrict__ Et,
                          const float* __restrict__ Wsm, const float* __restrict__ bsm,
                          const float* __restrict__ Wrm, const float* __restrict__ brm,
                          const float* __restrict__ Wam, const float* __restrict__ bam,
                          const float* __restrict__ g0, const float* __restrict__ b0,
                          float* __restrict__ xf, half_t* __restrict__ xh) {
  int srow = blockIdx.x;          // 0..SQ-1
  int b = blockIdx.y;
  int type = srow % 3, l = srow / 3;
  int tok = tt[b * LT + l];
  const float* pos = Et + (size_t)tok * DM;
  __shared__ float rowbuf[DM];
  __shared__ float sred[8];

  for (int d = threadIdx.x; d < DM; d += 256) {
    float e = pos[d];
    if (type == 0) {
      e += R[b * LT + l] * Wrm[d] + brm[d];
    } else if (type == 1) {
      float acc = 0.f;
      const float* sp = sv + (size_t)(b * LT + l) * STATE;
#pragma unroll
      for (int i = 0; i < STATE; ++i) acc += sp[i] * Wsm[i * DM + d];
      e += acc + bsm[d];
    } else {
      float acc = 0.f;
      const float* ap = av + (size_t)(b * LT + l) * ACT;
#pragma unroll
      for (int i = 0; i < ACT; ++i) acc += ap[i] * Wam[i * DM + d];
      e += acc + bam[d];
    }
    rowbuf[d] = e;
  }
  __syncthreads();

  float pacc = 0.f;
  for (int d = threadIdx.x; d < DM; d += 256) pacc += rowbuf[d];
  float mean = block_sum(pacc, sred) * (1.0f / DM);
  float qacc = 0.f;
  for (int d = threadIdx.x; d < DM; d += 256) {
    float dd = rowbuf[d] - mean;
    qacc += dd * dd;
  }
  float inv = rsqrtf(block_sum(qacc, sred) * (1.0f / DM) + 1e-5f);
  size_t o = ((size_t)b * SQ + srow) * DM;
  for (int d = threadIdx.x; d < DM; d += 256) {
    float vn = (rowbuf[d] - mean) * inv * g0[d] + b0[d];
    xf[o + d] = vn;
    xh[o + d] = (half_t)vn;
  }
}

// ---- generic WMMA GEMM: C[M,Nout] = A[M,K] * BT[Nout,K]^T (+bias,+relu) ---
__global__ void gemm_wmma_f16(const half_t* __restrict__ A,
                              const half_t* __restrict__ BT,
                              const float* __restrict__ bias,
                              half_t* __restrict__ outH, float* __restrict__ outF,
                              half_t* __restrict__ outT,   // [Nout, Sb] per batch
                              int M, int Nout, int K, int relu, int Sb) {
  int w = threadIdx.x >> 5;
  int lane = threadIdx.x & 31;
  int n = lane & 15, g = lane >> 4;
  int m_tile = blockIdx.y * 8 + w;
  int n_tile = blockIdx.x;
  if (m_tile * 16 >= M) return;

  const half_t* Arow = A + (size_t)m_tile * 16 * K;
  const half_t* Brow = BT + (size_t)n_tile * 16 * K;
  v8f acc = {};
  for (int k = 0; k < K; k += 32) {
    __builtin_prefetch(Arow + k + 128, 0, 1);
    __builtin_prefetch(Brow + k + 128, 0, 1);
    v16h a = load_frag(Arow + k, K);
    v16h b = load_frag(Brow + k, K);
    acc = wmma32(a, b, acc);
  }

  int col = n_tile * 16 + n;
  float bv = bias ? bias[col] : 0.0f;
#pragma unroll
  for (int r = 0; r < 8; ++r) {
    int row = m_tile * 16 + r + (g ? 8 : 0);
    float v = acc[r] + bv;
    if (relu) v = fmaxf(v, 0.0f);
    if (outH) outH[(size_t)row * Nout + col] = (half_t)v;
    if (outF) outF[(size_t)row * Nout + col] = v;
    if (outT) {
      int bb = row / Sb, ss = row - bb * Sb;
      outT[((size_t)bb * Nout + col) * Sb + ss] = (half_t)v;
    }
  }
}

// ---- flash attention: wave per 16-row Q tile, 32-key blocks ---------------
__global__ void flash_attn(const half_t* __restrict__ Q, const half_t* __restrict__ Kb,
                           const half_t* __restrict__ VT,   // [NB, H*HD, SQ]
                           half_t* __restrict__ O) {
  int w = threadIdx.x >> 5;
  int lane = threadIdx.x & 31;
  int n = lane & 15, g = lane >> 4;
  int qt = blockIdx.x * 8 + w;
  int h = blockIdx.y, b = blockIdx.z;
  int qbase = qt * 16;

  __shared__ __align__(16) half_t pbuf[8][16 * 34];
  half_t* pw = pbuf[w];

  const half_t* Qbase = Q + ((size_t)(b * SQ + qbase) * DM + h * HD);
  v16h aq0 = load_frag(Qbase, DM);
  v16h aq1 = load_frag(Qbase + 32, DM);

  v8f o0 = {}, o1 = {}, o2 = {}, o3 = {};
  float mrun[8], lrun[8];
#pragma unroll
  for (int r = 0; r < 8; ++r) { mrun[r] = -3.0e38f; lrun[r] = 0.0f; }
  const float scale = 0.125f;  // 1/sqrt(64)

  for (int k0 = 0; k0 <= qbase + 15; k0 += 32) {
    const half_t* Kbase = Kb + ((size_t)(b * SQ + k0) * DM + h * HD);
    v8f s0 = {}, s1 = {};
    s0 = wmma32(aq0, load_frag(Kbase, DM), s0);
    s0 = wmma32(aq1, load_frag(Kbase + 32, DM), s0);
    s1 = wmma32(aq0, load_frag(Kbase + 16 * DM, DM), s1);
    s1 = wmma32(aq1, load_frag(Kbase + 16 * DM + 32, DM), s1);

#pragma unroll
    for (int r = 0; r < 8; ++r) {
      int lrow = r + (g ? 8 : 0);
      int row = qbase + lrow;
      float v0 = s0[r] * scale; if (k0 + n > row)      v0 = -3.0e38f;
      float v1 = s1[r] * scale; if (k0 + 16 + n > row) v1 = -3.0e38f;
      float t = fmaxf(v0, v1);
#pragma unroll
      for (int msk = 8; msk >= 1; msk >>= 1) t = fmaxf(t, __shfl_xor(t, msk, 32));
      float mt = fmaxf(mrun[r], t);
      float al = __expf(mrun[r] - mt);
      float p0 = __expf(v0 - mt);
      float p1 = __expf(v1 - mt);
      float sm = p0 + p1;
#pragma unroll
      for (int msk = 8; msk >= 1; msk >>= 1) sm += __shfl_xor(sm, msk, 32);
      lrun[r] = lrun[r] * al + sm;
      mrun[r] = mt;
      o0[r] *= al; o1[r] *= al; o2[r] *= al; o3[r] *= al;
      pw[lrow * 34 + n]      = (half_t)p0;
      pw[lrow * 34 + 16 + n] = (half_t)p1;
    }
    // wave-private LDS transpose of P: make DS stores visible before reload
    asm volatile("s_wait_dscnt 0" ::: "memory");
    v16h pf = load_frag(pw, 34);

    const half_t* Vb = VT + ((size_t)b * DM + h * HD) * SQ + k0;
    o0 = wmma32(pf, load_frag(Vb + 0 * 16 * SQ, SQ), o0);
    o1 = wmma32(pf, load_frag(Vb + 1 * 16 * SQ, SQ), o1);
    o2 = wmma32(pf, load_frag(Vb + 2 * 16 * SQ, SQ), o2);
    o3 = wmma32(pf, load_frag(Vb + 3 * 16 * SQ, SQ), o3);
  }

#pragma unroll
  for (int r = 0; r < 8; ++r) {
    float inv = 1.0f / lrun[r];
    int row = qbase + r + (g ? 8 : 0);
    size_t ob = (size_t)(b * SQ + row) * DM + h * HD + n;
    O[ob +  0] = (half_t)(o0[r] * inv);
    O[ob + 16] = (half_t)(o1[r] * inv);
    O[ob + 32] = (half_t)(o2[r] * inv);
    O[ob + 48] = (half_t)(o3[r] * inv);
  }
}

// ---- residual add + layernorm ---------------------------------------------
__global__ void residual_ln(float* __restrict__ xf, half_t* __restrict__ xh,
                            const float* __restrict__ y,
                            const float* __restrict__ gam, const float* __restrict__ bet) {
  int row = blockIdx.x;
  __shared__ float rowbuf[DM];
  __shared__ float sred[8];
  size_t o = (size_t)row * DM;
  for (int d = threadIdx.x; d < DM; d += 256) rowbuf[d] = xf[o + d] + y[o + d];
  __syncthreads();
  float pacc = 0.f;
  for (int d = threadIdx.x; d < DM; d += 256) pacc += rowbuf[d];
  float mean = block_sum(pacc, sred) * (1.0f / DM);
  float qacc = 0.f;
  for (int d = threadIdx.x; d < DM; d += 256) {
    float dd = rowbuf[d] - mean;
    qacc += dd * dd;
  }
  float inv = rsqrtf(block_sum(qacc, sred) * (1.0f / DM) + 1e-5f);
  for (int d = threadIdx.x; d < DM; d += 256) {
    float vn = (rowbuf[d] - mean) * inv * gam[d] + bet[d];
    xf[o + d] = vn;
    xh[o + d] = (half_t)vn;
  }
}

// ---- final head: state tokens @ Wpa + bpa ---------------------------------
__global__ void final_proj(const float* __restrict__ xf, const float* __restrict__ Wpa,
                           const float* __restrict__ bpa, float* __restrict__ out) {
  int w = threadIdx.x >> 5, lane = threadIdx.x & 31;
  int tokidx = blockIdx.x * 8 + w;
  if (tokidx >= NB * LT) return;
  int b = tokidx / LT, l = tokidx - b * LT;
  const float* xr = xf + ((size_t)b * SQ + 3 * l + 1) * DM;
  float acc[ACT];
#pragma unroll
  for (int j = 0; j < ACT; ++j) acc[j] = 0.f;
  for (int d = lane; d < DM; d += 32) {
    float xv = xr[d];
#pragma unroll
    for (int j = 0; j < ACT; ++j) acc[j] += xv * Wpa[d * ACT + j];
  }
#pragma unroll
  for (int j = 0; j < ACT; ++j) {
#pragma unroll
    for (int m = 16; m >= 1; m >>= 1) acc[j] += __shfl_xor(acc[j], m, 32);
  }
  if (lane == 0) {
#pragma unroll
    for (int j = 0; j < ACT; ++j) out[(size_t)tokidx * ACT + j] = acc[j] + bpa[j];
  }
}

// ---------------------------------------------------------------------------
extern "C" void kernel_launch(void* const* d_in, const int* in_sizes, int n_in,
                              void* d_out, int out_size, void* d_ws, size_t ws_size,
                              hipStream_t stream) {
  (void)in_sizes; (void)n_in; (void)out_size; (void)ws_size;
  const float* R    = (const float*)d_in[0];
  const float* sv   = (const float*)d_in[1];
  const float* av   = (const float*)d_in[2];
  const int*   tt   = (const int*)d_in[3];
  const float* Et   = (const float*)d_in[4];
  const float* Wsm  = (const float*)d_in[5];
  const float* bsm  = (const float*)d_in[6];
  const float* Wrm  = (const float*)d_in[7];
  const float* brm  = (const float*)d_in[8];
  const float* Wam  = (const float*)d_in[9];
  const float* bam  = (const float*)d_in[10];
  const float* ln0g = (const float*)d_in[11];
  const float* ln0b = (const float*)d_in[12];
  const float* Wq   = (const float*)d_in[13];
  const float* Wk   = (const float*)d_in[14];
  const float* Wv   = (const float*)d_in[15];
  const float* Wo   = (const float*)d_in[16];
  const float* ln1g = (const float*)d_in[17];
  const float* ln1b = (const float*)d_in[18];
  const float* W1   = (const float*)d_in[19];
  const float* b1   = (const float*)d_in[20];
  const float* W2   = (const float*)d_in[21];
  const float* b2   = (const float*)d_in[22];
  const float* ln2g = (const float*)d_in[23];
  const float* ln2b = (const float*)d_in[24];
  const float* Wpa  = (const float*)d_in[25];
  const float* bpa  = (const float*)d_in[26];
  float* out = (float*)d_out;

  const size_t NS = (size_t)NB * SQ;  // 6144 rows
  char* p = (char*)d_ws;
  auto carve = [&](size_t bytes) -> void* {
    void* r = (void*)p;
    p += (bytes + 255) & ~(size_t)255;
    return r;
  };
  float*  xf  = (float*) carve(NS * DM * 4);
  half_t* xh  = (half_t*)carve(NS * DM * 2);
  float*  yf  = (float*) carve(NS * DM * 4);
  half_t* qh  = (half_t*)carve(NS * DM * 2);
  half_t* kh  = (half_t*)carve(NS * DM * 2);
  half_t* vt  = (half_t*)carve(NS * DM * 2);   // [NB, H*HD, SQ]
  half_t* at  = (half_t*)carve(NS * DM * 2);
  half_t* ffh = (half_t*)carve(NS * FFD * 2);
  half_t* WqT = (half_t*)carve((size_t)DM * DM * 2);
  half_t* WkT = (half_t*)carve((size_t)DM * DM * 2);
  half_t* WvT = (half_t*)carve((size_t)DM * DM * 2);
  half_t* WoT = (half_t*)carve((size_t)DM * DM * 2);
  half_t* W1T = (half_t*)carve((size_t)FFD * DM * 2);
  half_t* W2T = (half_t*)carve((size_t)DM * FFD * 2);

  dim3 blk(256);
  auto tg = [](int total) { return dim3((unsigned)((total + 255) / 256)); };

  embed_ln0<<<dim3(SQ, NB), blk, 0, stream>>>(R, sv, av, tt, Et, Wsm, bsm, Wrm, brm,
                                              Wam, bam, ln0g, ln0b, xf, xh);

  for (int l = 0; l < NLAY; ++l) {
    conv_transpose_f16<<<tg(DM * DM), blk, 0, stream>>>(Wq + (size_t)l * DM * DM, WqT, DM, DM);
    conv_transpose_f16<<<tg(DM * DM), blk, 0, stream>>>(Wk + (size_t)l * DM * DM, WkT, DM, DM);
    conv_transpose_f16<<<tg(DM * DM), blk, 0, stream>>>(Wv + (size_t)l * DM * DM, WvT, DM, DM);
    conv_transpose_f16<<<tg(DM * DM), blk, 0, stream>>>(Wo + (size_t)l * DM * DM, WoT, DM, DM);
    conv_transpose_f16<<<tg(DM * FFD), blk, 0, stream>>>(W1 + (size_t)l * DM * FFD, W1T, DM, FFD);
    conv_transpose_f16<<<tg(FFD * DM), blk, 0, stream>>>(W2 + (size_t)l * FFD * DM, W2T, FFD, DM);

    dim3 gqkv(DM / 16, (unsigned)(NS / 128));
    gemm_wmma_f16<<<gqkv, blk, 0, stream>>>(xh, WqT, nullptr, qh, nullptr, nullptr,
                                            (int)NS, DM, DM, 0, SQ);
    gemm_wmma_f16<<<gqkv, blk, 0, stream>>>(xh, WkT, nullptr, kh, nullptr, nullptr,
                                            (int)NS, DM, DM, 0, SQ);
    gemm_wmma_f16<<<gqkv, blk, 0, stream>>>(xh, WvT, nullptr, nullptr, nullptr, vt,
                                            (int)NS, DM, DM, 0, SQ);

    flash_attn<<<dim3(SQ / 16 / 8, NH, NB), blk, 0, stream>>>(qh, kh, vt, at);

    gemm_wmma_f16<<<gqkv, blk, 0, stream>>>(at, WoT, nullptr, nullptr, yf, nullptr,
                                            (int)NS, DM, DM, 0, SQ);
    residual_ln<<<dim3((unsigned)NS), blk, 0, stream>>>(xf, xh, yf,
                                                        ln1g + (size_t)l * DM,
                                                        ln1b + (size_t)l * DM);

    gemm_wmma_f16<<<dim3(FFD / 16, (unsigned)(NS / 128)), blk, 0, stream>>>(
        xh, W1T, b1 + (size_t)l * FFD, ffh, nullptr, nullptr, (int)NS, FFD, DM, 1, SQ);
    gemm_wmma_f16<<<gqkv, blk, 0, stream>>>(ffh, W2T, b2 + (size_t)l * DM, nullptr, yf,
                                            nullptr, (int)NS, DM, FFD, 0, SQ);
    residual_ln<<<dim3((unsigned)NS), blk, 0, stream>>>(xf, xh, yf,
                                                        ln2g + (size_t)l * DM,
                                                        ln2b + (size_t)l * DM);
  }

  final_proj<<<dim3(NB * LT / 8), blk, 0, stream>>>(xf, Wpa, bpa, out);
}
